// MultiHeadAttention_12120397709593
// MI455X (gfx1250) — matrix operand order repaired and checked by reading, not compiled, not run on hardware
//
#include <hip/hip_runtime.h>

// ---------------------------------------------------------------------------
// Types for CDNA5 WMMA (wave32): v_wmma_f32_16x16x32_bf16
// ---------------------------------------------------------------------------
typedef __attribute__((ext_vector_type(16))) __bf16 v16bf;
typedef __attribute__((ext_vector_type(8)))  float  v8f;

union BFrag {
    uint4 q[2];     // 32 bytes
    v16bf v;        // 16 bf16
};

static __device__ __forceinline__ unsigned short f32_to_bf16(float f) {
    unsigned int u = __float_as_uint(f);
    u = (u + 0x7FFFu + ((u >> 16) & 1u)) >> 16;   // round-to-nearest-even
    return (unsigned short)u;
}

static __device__ __forceinline__ v8f bf16_wmma(v16bf a, v16bf b, v8f c) {
    return __builtin_amdgcn_wmma_f32_16x16x32_bf16(
        /*neg_a=*/false, a, /*neg_b=*/false, b,
        /*c_mod=*/(short)0, c, /*reuse_a=*/false, /*reuse_b=*/false);
}

// A fragment (16xK=32, bf16), source row-major [rows x ld], tile at (row0,k0).
// ISA layout: lane m=l&15 holds row m; lanes 0-15: K {kb..kb+7, kb+16..kb+23}
// with kb = (l>>4)*8.
static __device__ __forceinline__ v16bf load_frag_A(const unsigned short* base,
                                                    int ld, int row0, int k0,
                                                    int lane) {
    int m  = lane & 15;
    int kb = (lane >> 4) * 8;
    const unsigned short* p = base + (size_t)(row0 + m) * ld + k0 + kb;
    BFrag f;
    f.q[0] = *(const uint4*)(p);
    f.q[1] = *(const uint4*)(p + 16);
    return f.v;
}

// Same layout but reading from LDS.
static __device__ __forceinline__ v16bf load_frag_A_lds(const unsigned short* base,
                                                        int lane) {
    int m  = lane & 15;
    int kb = (lane >> 4) * 8;
    const unsigned short* p = base + m * 32 + kb;
    BFrag f;
    f.q[0] = *(const uint4*)(p);
    f.q[1] = *(const uint4*)(p + 16);
    return f.v;
}

// B fragment (K=32 x 16, bf16) from row-major N x K storage (i.e. B^T rows).
// ISA layout: lane holds col n=l&15; K {kb..kb+7, kb+16..kb+23}
// with kb = (l>>4)*8 (same K order as the A fragment).
static __device__ __forceinline__ v16bf load_frag_B(const unsigned short* base,
                                                    int ld, int n0, int k0,
                                                    int lane) {
    int n  = lane & 15;
    int kk = (lane >> 4) * 8;
    const unsigned short* p = base + (size_t)(n0 + n) * ld + k0 + kk;
    BFrag f;
    f.q[0] = *(const uint4*)(p);
    f.q[1] = *(const uint4*)(p + 16);
    return f.v;
}

// ---------------------------------------------------------------------------
// Kernel 1: fp32 -> bf16 elementwise cast (x)
// ---------------------------------------------------------------------------
__global__ void cast_bf16_kernel(const float* __restrict__ x,
                                 unsigned short* __restrict__ xb, int n) {
    int i = blockIdx.x * blockDim.x + threadIdx.x;
    if (i < n) xb[i] = f32_to_bf16(x[i]);
}

// ---------------------------------------------------------------------------
// Kernel 2: transpose + cast weight [K,N] f32 -> [N,K] bf16 (LDS tiled)
// ---------------------------------------------------------------------------
__global__ void transpose_cast_kernel(const float* __restrict__ w,
                                      unsigned short* __restrict__ wt,
                                      int K, int N) {
    __shared__ float tile[32][33];
    int k0 = blockIdx.y * 32, n0 = blockIdx.x * 32;
    int tx = threadIdx.x, ty = threadIdx.y;           // 32 x 8
    #pragma unroll
    for (int yy = ty; yy < 32; yy += 8)
        tile[yy][tx] = w[(size_t)(k0 + yy) * N + n0 + tx];
    __syncthreads();
    #pragma unroll
    for (int yy = ty; yy < 32; yy += 8)
        wt[(size_t)(n0 + yy) * K + k0 + tx] = f32_to_bf16(tile[tx][yy]);
}

// ---------------------------------------------------------------------------
// Kernel 3: QKV GEMM. [4096,1024] @ [1024,3072] + bias.
// 128 threads = 4 waves; block tile 64(M) x 64(N); wave = 16(M) x 64(N).
// Double-buffered fragments so next-k loads overlap current-k WMMAs.
// Scatters results into K[B,H,T,D], Q[B,H,T,D], V^T[B,H,D,T] (bf16).
// ---------------------------------------------------------------------------
__global__ __launch_bounds__(128) void qkv_gemm_kernel(
        const unsigned short* __restrict__ xb,
        const unsigned short* __restrict__ w1t,   // [3072,1024] bf16 (N x K)
        const float* __restrict__ bias,           // [3072]
        unsigned short* __restrict__ Kb,
        unsigned short* __restrict__ Qb,
        unsigned short* __restrict__ Vt) {
    int lane = threadIdx.x & 31;
    int w    = threadIdx.x >> 5;
    int mBase = blockIdx.y * 64 + w * 16;
    int nBase = blockIdx.x * 64;

    v8f acc[4] = {};

    v16bf a  = load_frag_A(xb, 1024, mBase, 0, lane);
    v16bf b0 = load_frag_B(w1t, 1024, nBase,      0, lane);
    v16bf b1 = load_frag_B(w1t, 1024, nBase + 16, 0, lane);
    v16bf b2 = load_frag_B(w1t, 1024, nBase + 32, 0, lane);
    v16bf b3 = load_frag_B(w1t, 1024, nBase + 48, 0, lane);

    for (int k0 = 0; k0 < 1024; k0 += 32) {
        int kn = (k0 + 32 < 1024) ? (k0 + 32) : k0;   // last iter: harmless reload
        v16bf an  = load_frag_A(xb, 1024, mBase, kn, lane);
        v16bf bn0 = load_frag_B(w1t, 1024, nBase,      kn, lane);
        v16bf bn1 = load_frag_B(w1t, 1024, nBase + 16, kn, lane);
        v16bf bn2 = load_frag_B(w1t, 1024, nBase + 32, kn, lane);
        v16bf bn3 = load_frag_B(w1t, 1024, nBase + 48, kn, lane);

        acc[0] = bf16_wmma(a, b0, acc[0]);
        acc[1] = bf16_wmma(a, b1, acc[1]);
        acc[2] = bf16_wmma(a, b2, acc[2]);
        acc[3] = bf16_wmma(a, b3, acc[3]);

        a = an; b0 = bn0; b1 = bn1; b2 = bn2; b3 = bn3;
    }

    int g = lane >> 4;
    #pragma unroll
    for (int s = 0; s < 4; ++s) {
        int n   = nBase + s * 16 + (lane & 15);
        float bv = bias[n];
        int sel = n >> 10;          // 0:K 1:Q 2:V
        int c   = n & 1023;
        int h   = c >> 6, d = c & 63;
        #pragma unroll
        for (int r = 0; r < 8; ++r) {
            int m  = mBase + r + 8 * g;
            int b_ = m >> 11, t = m & 2047;
            unsigned short v16 = f32_to_bf16(acc[s][r] + bv);
            size_t bh = (size_t)(b_ * 16 + h);
            if (sel == 0)      Kb[(bh * 2048 + t) * 64 + d] = v16;
            else if (sel == 1) Qb[(bh * 2048 + t) * 64 + d] = v16;
            else               Vt[(bh * 64 + d) * 2048 + t] = v16;
        }
    }
}

// ---------------------------------------------------------------------------
// Kernel 4: flash attention per (b,h). Reference computes wei = K @ Q^T,
// softmax over j, out = wei @ V  => "query" role is played by K rows.
// Block: 4 waves x 16 rows = 64 rows of i per block. j streamed in steps of 32.
// V loads hoisted above the softmax; next Q tile prefetched before softmax.
// ---------------------------------------------------------------------------
__global__ __launch_bounds__(128) void attn_kernel(
        const unsigned short* __restrict__ Kb,
        const unsigned short* __restrict__ Qb,
        const unsigned short* __restrict__ Vt,
        unsigned short* __restrict__ ao) {      // [B,T,H,D] bf16
    __shared__ __align__(16) unsigned short lp[4][16 * 32];   // P tile per wave

    int lane = threadIdx.x & 31;
    int w    = threadIdx.x >> 5;
    int bh   = blockIdx.y;                      // b*16 + h
    int i0   = blockIdx.x * 64 + w * 16;
    int g    = lane >> 4;
    int nl   = lane & 15;

    const unsigned short* Kh = Kb + (size_t)bh * 2048 * 64;
    const unsigned short* Qh = Qb + (size_t)bh * 2048 * 64;
    const unsigned short* Vh = Vt + (size_t)bh * 64 * 2048;

    // Resident "query" fragments: rows i0..i0+15 of K, D split into 2x K32.
    v16bf aK0 = load_frag_A(Kh, 64, i0, 0, lane);
    v16bf aK1 = load_frag_A(Kh, 64, i0, 32, lane);

    v8f o[4] = {};
    float mrun[8], lrun[8];
    #pragma unroll
    for (int r = 0; r < 8; ++r) { mrun[r] = -1e30f; lrun[r] = 0.0f; }

    int jend = i0 + 16;   // exclusive bound on jb (causal)

    v16bf q00 = load_frag_B(Qh, 64, 0,  0,  lane);
    v16bf q01 = load_frag_B(Qh, 64, 0,  32, lane);
    v16bf q10 = load_frag_B(Qh, 64, 16, 0,  lane);
    v16bf q11 = load_frag_B(Qh, 64, 16, 32, lane);

    for (int jb = 0; jb < jend; jb += 32) {
        // Scores: S[i, jb..jb+31] = K_i . Q_j / 8
        v8f s0 = {}, s1 = {};
        s0 = bf16_wmma(aK0, q00, s0);
        s0 = bf16_wmma(aK1, q01, s0);
        s1 = bf16_wmma(aK0, q10, s1);
        s1 = bf16_wmma(aK1, q11, s1);

        // hoist V fragment loads: consumed only after the softmax below
        v16bf vB0 = load_frag_B(Vh, 2048, 0,  jb, lane);
        v16bf vB1 = load_frag_B(Vh, 2048, 16, jb, lane);
        v16bf vB2 = load_frag_B(Vh, 2048, 32, jb, lane);
        v16bf vB3 = load_frag_B(Vh, 2048, 48, jb, lane);

        // prefetch next j-tile's Q fragments (last iter: reload current tile)
        int jn = (jb + 32 < jend) ? (jb + 32) : jb;
        q00 = load_frag_B(Qh, 64, jn,      0,  lane);
        q01 = load_frag_B(Qh, 64, jn,      32, lane);
        q10 = load_frag_B(Qh, 64, jn + 16, 0,  lane);
        q11 = load_frag_B(Qh, 64, jn + 16, 32, lane);

        int j0 = jb + nl, j1 = jb + 16 + nl;
        #pragma unroll
        for (int r = 0; r < 8; ++r) {
            int i = i0 + r + 8 * g;
            float v0 = (j0 <= i) ? s0[r] * 0.125f : -1e30f;
            float v1 = (j1 <= i) ? s1[r] * 0.125f : -1e30f;

            // row max over the 16 lanes holding this row's columns
            float mx = fmaxf(v0, v1);
            mx = fmaxf(mx, __shfl_xor(mx, 1));
            mx = fmaxf(mx, __shfl_xor(mx, 2));
            mx = fmaxf(mx, __shfl_xor(mx, 4));
            mx = fmaxf(mx, __shfl_xor(mx, 8));
            float mnew  = fmaxf(mrun[r], mx);
            float alpha = __expf(mrun[r] - mnew);
            float p0 = __expf(v0 - mnew);
            float p1 = __expf(v1 - mnew);
            float rs = p0 + p1;
            rs += __shfl_xor(rs, 1);
            rs += __shfl_xor(rs, 2);
            rs += __shfl_xor(rs, 4);
            rs += __shfl_xor(rs, 8);
            lrun[r] = lrun[r] * alpha + rs;
            mrun[r] = mnew;
            #pragma unroll
            for (int s = 0; s < 4; ++s) o[s][r] *= alpha;

            // stage P (bf16) into LDS in plain row-major 16x32
            int row = r + 8 * g;
            lp[w][row * 32 + nl]      = f32_to_bf16(p0);
            lp[w][row * 32 + 16 + nl] = f32_to_bf16(p1);
        }
        asm volatile("s_wait_dscnt 0" ::: "memory");

        // reload P as a WMMA A-fragment (K = 32 j-columns)
        v16bf pA = load_frag_A_lds(lp[w], lane);

        // O += P @ V : V^T rows are contiguous along j
        o[0] = bf16_wmma(pA, vB0, o[0]);
        o[1] = bf16_wmma(pA, vB1, o[1]);
        o[2] = bf16_wmma(pA, vB2, o[2]);
        o[3] = bf16_wmma(pA, vB3, o[3]);
    }

    // epilogue: normalize and write [B,T,H,D] bf16
    int b_ = bh >> 4, h = bh & 15;
    #pragma unroll
    for (int r = 0; r < 8; ++r) {
        float inv = 1.0f / lrun[r];
        int i = i0 + r + 8 * g;
        size_t rowbase = ((size_t)(b_ * 2048 + i)) * 1024 + h * 64;
        #pragma unroll
        for (int s = 0; s < 4; ++s)
            ao[rowbase + s * 16 + nl] = f32_to_bf16(o[s][r] * inv);
    }
}

// ---------------------------------------------------------------------------
// Kernel 5: output projection GEMM [4096,1024] @ [1024,1024] + bias -> f32
// Same double-buffered structure as the QKV GEMM.
// ---------------------------------------------------------------------------
__global__ __launch_bounds__(128) void proj_gemm_kernel(
        const unsigned short* __restrict__ ab,    // [4096,1024] bf16
        const unsigned short* __restrict__ w2t,   // [1024,1024] bf16 (N x K)
        const float* __restrict__ bias,
        float* __restrict__ out) {
    int lane = threadIdx.x & 31;
    int w    = threadIdx.x >> 5;
    int mBase = blockIdx.y * 64 + w * 16;
    int nBase = blockIdx.x * 64;

    v8f acc[4] = {};

    v16bf a  = load_frag_A(ab, 1024, mBase, 0, lane);
    v16bf b0 = load_frag_B(w2t, 1024, nBase,      0, lane);
    v16bf b1 = load_frag_B(w2t, 1024, nBase + 16, 0, lane);
    v16bf b2 = load_frag_B(w2t, 1024, nBase + 32, 0, lane);
    v16bf b3 = load_frag_B(w2t, 1024, nBase + 48, 0, lane);

    for (int k0 = 0; k0 < 1024; k0 += 32) {
        int kn = (k0 + 32 < 1024) ? (k0 + 32) : k0;
        v16bf an  = load_frag_A(ab, 1024, mBase, kn, lane);
        v16bf bn0 = load_frag_B(w2t, 1024, nBase,      kn, lane);
        v16bf bn1 = load_frag_B(w2t, 1024, nBase + 16, kn, lane);
        v16bf bn2 = load_frag_B(w2t, 1024, nBase + 32, kn, lane);
        v16bf bn3 = load_frag_B(w2t, 1024, nBase + 48, kn, lane);

        acc[0] = bf16_wmma(a, b0, acc[0]);
        acc[1] = bf16_wmma(a, b1, acc[1]);
        acc[2] = bf16_wmma(a, b2, acc[2]);
        acc[3] = bf16_wmma(a, b3, acc[3]);

        a = an; b0 = bn0; b1 = bn1; b2 = bn2; b3 = bn3;
    }

    int g = lane >> 4;
    #pragma unroll
    for (int s = 0; s < 4; ++s) {
        int n   = nBase + s * 16 + (lane & 15);
        float bv = bias[n];
        #pragma unroll
        for (int r = 0; r < 8; ++r) {
            int m = mBase + r + 8 * g;
            out[(size_t)m * 1024 + n] = acc[s][r] + bv;
        }
    }
}

// ---------------------------------------------------------------------------
// Host launcher
// ---------------------------------------------------------------------------
extern "C" void kernel_launch(void* const* d_in, const int* in_sizes, int n_in,
                              void* d_out, int out_size, void* d_ws, size_t ws_size,
                              hipStream_t stream) {
    const float* x  = (const float*)d_in[0];   // [2,2048,1024]
    const float* w1 = (const float*)d_in[1];   // [1024,3072]
    const float* b1 = (const float*)d_in[2];   // [3072]
    const float* w2 = (const float*)d_in[3];   // [1024,1024]
    const float* b2 = (const float*)d_in[4];   // [1024]
    float* out = (float*)d_out;

    char* ws = (char*)d_ws;
    const size_t MB = 1u << 20;
    unsigned short* xb  = (unsigned short*)(ws);             //  8 MiB: x bf16
    unsigned short* w1t = (unsigned short*)(ws + 8  * MB);   //  6 MiB: W1^T bf16
    unsigned short* w2t = (unsigned short*)(ws + 14 * MB);   //  2 MiB: W2^T bf16
    unsigned short* Kb  = (unsigned short*)(ws + 16 * MB);   //  8 MiB: K [B,H,T,D]
    unsigned short* Qb  = (unsigned short*)(ws + 24 * MB);   //  8 MiB: Q [B,H,T,D]
    unsigned short* Vt  = (unsigned short*)(ws + 32 * MB);   //  8 MiB: V^T [B,H,D,T]
    unsigned short* ao  = (unsigned short*)(ws + 40 * MB);   //  8 MiB: attn out [B,T,C]

    const int BT = 2 * 2048;   // 4096 rows

    cast_bf16_kernel<<<(BT * 1024) / 256, 256, 0, stream>>>(x, xb, BT * 1024);
    transpose_cast_kernel<<<dim3(3072 / 32, 1024 / 32), dim3(32, 8), 0, stream>>>(
        w1, w1t, 1024, 3072);
    transpose_cast_kernel<<<dim3(1024 / 32, 1024 / 32), dim3(32, 8), 0, stream>>>(
        w2, w2t, 1024, 1024);

    qkv_gemm_kernel<<<dim3(3072 / 64, BT / 64), 128, 0, stream>>>(
        xb, w1t, b1, Kb, Qb, Vt);

    attn_kernel<<<dim3(2048 / 64, 32), 128, 0, stream>>>(Kb, Qb, Vt, ao);

    proj_gemm_kernel<<<dim3(1024 / 64, BT / 64), 128, 0, stream>>>(
        ao, w2t, b2, out);
}
